// SparseSelfAttention_3040836845874
// MI455X (gfx1250) — compile-verified, matrix operand order checked by reading, not run on hardware
//
#include <hip/hip_runtime.h>

// ---------------- problem constants ----------------
#define BSZ    2
#define NH     16
#define SEQ    4096
#define DIM    64
#define BLK    64
#define NGLB   256          // 4 global positions per block * 64 blocks
#define QC     32           // query rows per workgroup
#define NCOL   (BLK + NGLB) // 320 score columns per row (class-major global order)
#define NEGV   (-10000.0f)

typedef __attribute__((ext_vector_type(16))) __bf16 v16bf;
typedef __attribute__((ext_vector_type(8)))  float  v8f;
typedef unsigned int u32x4 __attribute__((ext_vector_type(4)));
typedef int          i32x8 __attribute__((ext_vector_type(8)));
typedef int          i32x4 __attribute__((ext_vector_type(4)));

union FragBF { v16bf v; uint4 q[2]; };

__device__ __forceinline__ unsigned short f2bf(float f) {
  unsigned u = __float_as_uint(f);
  return (unsigned short)((u + 0x7FFFu + ((u >> 16) & 1u)) >> 16);  // RNE
}

// ---------------- LDS pool (manual offsets so TDM staging can overlay Sc) ----
#define QS_OFF   0u          //  4 KB bf16 Q       32x64
#define KC_OFF   4096u       // 40 KB bf16 K-cat  320x64 (0..63 local, then class0,61,62,63)
#define VT_OFF   45056u      // 40 KB bf16 V-cat^T 64x320
#define SC_OFF   86016u      // 40 KB f32 scores  32x320  (also TDM raw-f32 staging)
#define PB_OFF   126976u     // 20 KB bf16 probs  32x320
#define MB_OFF   147456u     // 1.25 KB f32 mask bias
#define RED_OFF  148736u     //  2 KB reduction scratch
#define POOL_SZ  150784u

// ---------------- Tensor Data Mover: 2D tile -> LDS ----------------
// D# built per CDNA5 ISA: group0 {count,lds,global,type=2},
// group1 {data_size=4B, tensor_dim0=64, tensor_dim1=rows, tile 64 x rows,
//         tensor_dim0_stride = row_stride (elements)}, groups 2/3 zero (2D).
__device__ __forceinline__ void tdm_load_2d(const float* gptr, unsigned lds_off,
                                            int rows, int row_stride_elems) {
  unsigned long long ga = (unsigned long long)(const void*)gptr;
  u32x4 g0;
  g0[0] = 1u;                                               // count=1 user descriptor
  g0[1] = lds_off;                                          // LDS byte address
  g0[2] = (unsigned)ga;                                     // global addr [31:0]
  g0[3] = (unsigned)((ga >> 32) & 0x1FFFFFFu) | (2u << 30); // addr [56:32] | type=2
  i32x8 g1;
  g1[0] = (2 << 16);                  // wg_mask=0 | data_size=4B | no flags
  g1[1] = (DIM & 0xFFFF) << 16;       // tensor_dim0[15:0] = 64
  g1[2] = (rows & 0xFFFF) << 16;      // tensor_dim0 hi=0 | tensor_dim1[15:0]
  g1[3] = (DIM & 0xFFFF) << 16;       // tensor_dim1 hi=0 | tile_dim0 = 64
  g1[4] = rows & 0xFFFF;              // tile_dim1 = rows | tile_dim2 = 0
  g1[5] = row_stride_elems;           // tensor_dim0_stride[31:0]
  g1[6] = 0;                          // stride0 hi | dim1_stride lo
  g1[7] = 0;
  i32x4 z4 = {0, 0, 0, 0};
  i32x8 z8 = {0, 0, 0, 0, 0, 0, 0, 0};
  // 6-arg toolchain form: (g0, g1, g2, g3, extra, cpol)
  __builtin_amdgcn_tensor_load_to_lds(g0, g1, z4, z4, z8, 0);
}

// convert a staged f32 tile (rows x 64, row-major) to bf16 rows of dst
__device__ __forceinline__ void conv_rows(const float* stg, unsigned short* dst,
                                          int base, int rows, int tid) {
  for (int idx = tid; idx < rows * 16; idx += 256) {
    int r = idx >> 4, c4 = (idx & 15) * 4;
    float4 f = *(const float4*)(stg + r * DIM + c4);
    unsigned short* d = &dst[(base + r) * DIM + c4];
    d[0] = f2bf(f.x); d[1] = f2bf(f.y); d[2] = f2bf(f.z); d[3] = f2bf(f.w);
  }
}

// convert a staged f32 tile (rows x 64) transposed into VT[64][NCOL]
__device__ __forceinline__ void conv_rows_T(const float* stg, unsigned short* VT,
                                            int base, int rows, int tid) {
  for (int idx = tid; idx < rows * 16; idx += 256) {
    int r = idx >> 4, c4 = (idx & 15) * 4;
    float4 f = *(const float4*)(stg + r * DIM + c4);
    int vrow = base + r;
    VT[(c4 + 0) * NCOL + vrow] = f2bf(f.x);
    VT[(c4 + 1) * NCOL + vrow] = f2bf(f.y);
    VT[(c4 + 2) * NCOL + vrow] = f2bf(f.z);
    VT[(c4 + 3) * NCOL + vrow] = f2bf(f.w);
  }
}

__global__ __launch_bounds__(256)
void sparse_attn_wmma(const float* __restrict__ q, const float* __restrict__ k,
                      const float* __restrict__ v, const float* __restrict__ mask,
                      float* __restrict__ out)
{
  __shared__ alignas(16) unsigned char pool[POOL_SZ];
  unsigned short* Qs = (unsigned short*)(pool + QS_OFF);
  unsigned short* Kc = (unsigned short*)(pool + KC_OFF);
  unsigned short* VT = (unsigned short*)(pool + VT_OFF);
  float*          Sc = (float*)(pool + SC_OFF);
  unsigned short* Pb = (unsigned short*)(pool + PB_OFF);
  float*          mb = (float*)(pool + MB_OFF);
  float*      redmax = (float*)(pool + RED_OFF);
  float*      redsum = redmax + QC * 8;

  const int tid  = threadIdx.x;
  const int lane = tid & 31;
  const int wave = tid >> 5;
  const int half = lane >> 4;   // half-wave K-chunk select (ISA 16-bit layouts)
  const int l16  = lane & 15;

  const int gid = blockIdx.x;
  const int qc  = gid & 127;            // SEQ/QC = 128 chunks
  const int h   = (gid >> 7) & 15;
  const int b   = gid >> 11;
  const int q0  = qc * QC;
  const int nb  = q0 / BLK;

  const size_t bh = ((size_t)b * NH + h) * SEQ;
  const float* qp = q + (bh + q0) * DIM;
  const float* kp = k + bh * DIM;
  const float* vp = v + bh * DIM;

  const float*   stg     = Sc;                        // raw-f32 TDM staging overlay
  const unsigned stg_lds = (unsigned)(size_t)(const void*)stg;
  const unsigned stg_hi  = stg_lds + 64u * DIM * 4u;  // second 16KB slot

  // mask bias, class-major global column order: col>=64 -> class c, block blk
  if (tid < NCOL) {
    int srow;
    if (tid < BLK) srow = nb * BLK + tid;
    else { int c = (tid - BLK) >> 6, blk = (tid - BLK) & 63;
           srow = blk * BLK + (c ? 60 + c : 0); }
    mb[tid] = (mask[(size_t)b * SEQ + srow] == 0.0f) ? NEGV : 0.0f;
  }

  // ---------------- TDM-staged load + bf16 convert ----------------
  // K chunk 0: local block + class0 (rows 0,64,128,... of k)
  if (wave == 0) {
    tdm_load_2d(kp + (size_t)nb * BLK * DIM, stg_lds, 64, DIM);
    tdm_load_2d(kp,                          stg_hi,  64, BLK * DIM);
    __builtin_amdgcn_s_wait_tensorcnt(0);
  }
  __syncthreads();
  conv_rows(stg, Kc, 0, 128, tid);
  __syncthreads();
  // K chunk 1: class 61 + class 62
  if (wave == 0) {
    tdm_load_2d(kp + 61 * DIM, stg_lds, 64, BLK * DIM);
    tdm_load_2d(kp + 62 * DIM, stg_hi,  64, BLK * DIM);
    __builtin_amdgcn_s_wait_tensorcnt(0);
  }
  __syncthreads();
  conv_rows(stg, Kc, 128, 128, tid);
  __syncthreads();
  // K chunk 2: class 63
  if (wave == 0) {
    tdm_load_2d(kp + 63 * DIM, stg_lds, 64, BLK * DIM);
    __builtin_amdgcn_s_wait_tensorcnt(0);
  }
  __syncthreads();
  conv_rows(stg, Kc, 256, 64, tid);
  __syncthreads();
  // V chunks (same shapes, converted transposed into VT)
  if (wave == 0) {
    tdm_load_2d(vp + (size_t)nb * BLK * DIM, stg_lds, 64, DIM);
    tdm_load_2d(vp,                          stg_hi,  64, BLK * DIM);
    __builtin_amdgcn_s_wait_tensorcnt(0);
  }
  __syncthreads();
  conv_rows_T(stg, VT, 0, 128, tid);
  __syncthreads();
  if (wave == 0) {
    tdm_load_2d(vp + 61 * DIM, stg_lds, 64, BLK * DIM);
    tdm_load_2d(vp + 62 * DIM, stg_hi,  64, BLK * DIM);
    __builtin_amdgcn_s_wait_tensorcnt(0);
  }
  __syncthreads();
  conv_rows_T(stg, VT, 128, 128, tid);
  __syncthreads();
  if (wave == 0) {
    tdm_load_2d(vp + 63 * DIM, stg_lds, 64, BLK * DIM);
    __builtin_amdgcn_s_wait_tensorcnt(0);
  }
  __syncthreads();
  conv_rows_T(stg, VT, 256, 64, tid);
  __syncthreads();
  // Q chunk
  if (wave == 0) {
    tdm_load_2d(qp, stg_lds, QC, DIM);
    __builtin_amdgcn_s_wait_tensorcnt(0);
  }
  __syncthreads();
  conv_rows(stg, Qs, 0, QC, tid);
  __syncthreads();

  // ---------------- phase 1: scores = Q @ Kcat^T  (WMMA bf16) ----------------
  for (int t = wave; t < 2 * 20; t += 8) {
    const int mt = t / 20, nt = t % 20;
    v8f acc = {};
#pragma unroll
    for (int kk = 0; kk < DIM; kk += 32) {
      FragBF a, bm;
      { const unsigned short* p = &Qs[(mt * 16 + l16) * DIM + kk + half * 8];
        a.q[0] = *(const uint4*)p;  a.q[1] = *(const uint4*)(p + 16); }
      { const unsigned short* p = &Kc[(nt * 16 + l16) * DIM + kk + half * 8];
        bm.q[0] = *(const uint4*)p; bm.q[1] = *(const uint4*)(p + 16); }
      acc = __builtin_amdgcn_wmma_f32_16x16x32_bf16(false, a.v, false, bm.v,
                                                    (short)0, acc, false, false);
    }
    const int col    = nt * 16 + l16;
    const float bias = mb[col];
    const int rbase  = mt * 16 + half * 8;     // D layout: M = elem + 8*half
#pragma unroll
    for (int j = 0; j < 8; ++j)
      Sc[(rbase + j) * NCOL + col] = acc[j] + bias;
  }
  __syncthreads();

  // ---------------- phase 2: f32 softmax over 320 cols ----------------
  {
    const int r = tid >> 3, part = tid & 7;
    const int c0 = part * 40;
    float m = -3.4e38f;
    for (int c = 0; c < 40; ++c) m = fmaxf(m, Sc[r * NCOL + c0 + c]);
    redmax[r * 8 + part] = m;
    __syncthreads();
    float rm = redmax[r * 8];
    for (int p = 1; p < 8; ++p) rm = fmaxf(rm, redmax[r * 8 + p]);
    float s = 0.0f;
    for (int c = 0; c < 40; ++c) {
      float e = __expf(Sc[r * NCOL + c0 + c] - rm);
      Sc[r * NCOL + c0 + c] = e;
      s += e;
    }
    redsum[r * 8 + part] = s;
    __syncthreads();
    float rs = 0.0f;
    for (int p = 0; p < 8; ++p) rs += redsum[r * 8 + p];
    const float inv = 1.0f / rs;
    for (int c = 0; c < 40; ++c)
      Pb[r * NCOL + c0 + c] = f2bf(Sc[r * NCOL + c0 + c] * inv);
  }
  __syncthreads();

  // ---------------- phase 3: out = P @ Vcat  (WMMA bf16, K = 320) ----------------
  {
    const int mt = wave >> 2, ntile = wave & 3;
    v8f acc = {};
#pragma unroll
    for (int kk = 0; kk < NCOL; kk += 32) {
      FragBF a, bm;
      { const unsigned short* p = &Pb[(mt * 16 + l16) * NCOL + kk + half * 8];
        a.q[0] = *(const uint4*)p;  a.q[1] = *(const uint4*)(p + 16); }
      { const unsigned short* p = &VT[(ntile * 16 + l16) * NCOL + kk + half * 8];
        bm.q[0] = *(const uint4*)p; bm.q[1] = *(const uint4*)(p + 16); }
      acc = __builtin_amdgcn_wmma_f32_16x16x32_bf16(false, a.v, false, bm.v,
                                                    (short)0, acc, false, false);
    }
    float* op = out + (bh + q0) * DIM;
    const int col   = ntile * 16 + l16;
    const int rbase = mt * 16 + half * 8;
#pragma unroll
    for (int j = 0; j < 8; ++j)
      op[(size_t)(rbase + j) * DIM + col] = acc[j];
  }
}

extern "C" void kernel_launch(void* const* d_in, const int* in_sizes, int n_in,
                              void* d_out, int out_size, void* d_ws, size_t ws_size,
                              hipStream_t stream) {
  const float* q    = (const float*)d_in[0];
  const float* k    = (const float*)d_in[1];
  const float* v    = (const float*)d_in[2];
  const float* mask = (const float*)d_in[3];
  float* out = (float*)d_out;

  const int grid = BSZ * NH * (SEQ / QC);   // 4096 workgroups
  sparse_attn_wmma<<<grid, 256, 0, stream>>>(q, k, v, mask, out);
}
